// PaiNN_20289425506991
// MI455X (gfx1250) — matrix operand order repaired and checked by reading, compiled, not run on hardware
//
#include <hip/hip_runtime.h>
#include <hip/hip_bf16.h>
#include <cstdint>

#define R_CUT    5.0f
#define N_NODES  50000
#define N_EDGES  400000
#define N_GRAPHS 2500
#define DD       128
#define N_RBF    20
#define PI_F     3.14159265358979f

typedef __attribute__((ext_vector_type(16))) _Float16 v16h;
typedef __attribute__((ext_vector_type(8)))  float    v8f;

// ---------------------------------------------------------------------------
// Pack B (K x N, f32 row-major) into per-lane WMMA layout, f16:
//   Bp[(((kc*(N/16) + ng)*32 + lane)*16 + h] = B[kc*32 + (lane>>4)*16 + h][ng*16 + (lane&15)]
// so each lane's 16 halves for one k-step are a contiguous 32B block.
// ---------------------------------------------------------------------------
__global__ void pack_b_kernel(const float* __restrict__ B, _Float16* __restrict__ Bp,
                              int K, int N) {
  const int idx = blockIdx.x * blockDim.x + threadIdx.x;
  if (idx >= K * N) return;
  const int h    = idx & 15;
  const int lane = (idx >> 4) & 31;
  const int rest = idx >> 9;
  const int ngrp = rest % (N >> 4);
  const int kc   = rest / (N >> 4);
  const int k = kc * 32 + (lane >> 4) * 16 + h;
  const int n = ngrp * 16 + (lane & 15);
  Bp[idx] = (_Float16)B[(size_t)k * N + n];
}

// ---------------------------------------------------------------------------
// Tall-skinny GEMM: C[M,N] = act(A[M,K] @ B[K,N] + bias[N]), f16 WMMA, f32 acc.
// Block = 256 threads = 8 waves covering 16 rows x 128 cols.
// A tile staged f32 -> LDS via gfx1250 async copy (ASYNCcnt); B pre-packed f16.
// ---------------------------------------------------------------------------
template <int ACT, int KT>
__global__ void __launch_bounds__(256)
wmma_gemm_kernel(const float* __restrict__ A, const _Float16* __restrict__ Bp,
                 const float* __restrict__ bias, float* __restrict__ C,
                 int M, int N) {
  constexpr int ldA = KT + 8;                  // floats; keeps 16B alignment
  __shared__ float As[16 * ldA];

  const int tid  = threadIdx.x;
  const int lane = tid & 31;
  const int wave = tid >> 5;
  const int m0   = blockIdx.x * 16;
  const int n0   = blockIdx.y * 128 + wave * 16;
  const int ngrp = blockIdx.y * 8 + wave;      // 16-col group index

  // ---- async-stage A tile (16 x KT f32) into LDS, 16B chunks ----
  constexpr int CPR = KT / 4;                  // b128 chunks per row
  constexpr int CHUNKS = 16 * CPR;
  for (int c = tid; c < CHUNKS; c += 256) {
    const int r  = c / CPR;
    const int kq = c - r * CPR;
    const int row = m0 + r;
    // wave-relative LDS byte offset = low 32 bits of flat shared address
    const unsigned lds_off = (unsigned)(uintptr_t)(const void*)&As[r * ldA + kq * 4];
    if (row < M) {
      const float* gsrc = A + (size_t)row * KT + (size_t)kq * 4;
      asm volatile("global_load_async_to_lds_b128 %0, %1, off"
                   :: "v"(lds_off), "v"(gsrc) : "memory");
    } else {
      As[r * ldA + kq * 4 + 0] = 0.0f;
      As[r * ldA + kq * 4 + 1] = 0.0f;
      As[r * ldA + kq * 4 + 2] = 0.0f;
      As[r * ldA + kq * 4 + 3] = 0.0f;
    }
  }
  asm volatile("s_wait_asynccnt 0x0" ::: "memory");
  __syncthreads();

  v8f acc = {};
  const int mloc = lane & 15;                  // A row within tile
  const int hi   = lane >> 4;                  // half-wave
  const int kbA  = hi * 16;
  const int ncol = n0 + (lane & 15);
  const int ngroups = N >> 4;

  #pragma unroll
  for (int kk = 0; kk < KT; kk += 32) {
    v16h a;
    #pragma unroll
    for (int h = 0; h < 8; ++h) {
      a[h]     = (_Float16)As[mloc * ldA + kk + kbA + h];
      a[h + 8] = (_Float16)As[mloc * ldA + kk + 8 + kbA + h];
    }
    // one contiguous 32B block per lane (2x global_load_b128, coalesced)
    const size_t boff = ((((size_t)(kk >> 5) * ngroups + ngrp) << 5) + lane) << 4;
    const v16h b = *(const v16h*)(Bp + boff);
    acc = __builtin_amdgcn_wmma_f32_16x16x32_f16(false, a, false, b,
                                                 (short)0, acc, false, false);
  }

  const float bval = bias ? bias[ncol] : 0.0f;
  #pragma unroll
  for (int r = 0; r < 8; ++r) {
    const int row = m0 + r + 8 * hi;
    if (row < M) {
      float val = acc[r] + bval;
      if (ACT) val = val / (1.0f + __expf(-val));   // SiLU
      C[(size_t)row * N + ncol] = val;
    }
  }
}

// ---------------------------------------------------------------------------
// Init: s = emb[z], v = 0
// ---------------------------------------------------------------------------
__global__ void init_nodes_kernel(const int* __restrict__ z,
                                  const float* __restrict__ emb,
                                  float* __restrict__ s, float* __restrict__ v,
                                  int n_nodes) {
  const int idx = blockIdx.x * blockDim.x + threadIdx.x;
  if (idx < n_nodes * 3 * DD) v[idx] = 0.0f;
  if (idx < n_nodes * DD) {
    const int node = idx >> 7, j = idx & 127;
    s[idx] = emb[z[node] * DD + j];
  }
}

// Per-edge RBF basis + cosine cutoff
__global__ void init_edges_kernel(const float* __restrict__ r_ij,
                                  float* __restrict__ rbf, float* __restrict__ fc,
                                  int n_edges) {
  const int e = blockIdx.x * blockDim.x + threadIdx.x;
  if (e >= n_edges) return;
  const float r = r_ij[e];
  fc[e] = (r <= R_CUT) ? 0.5f * (__cosf(PI_F * r / R_CUT) + 1.0f) : 0.0f;
  const float inv = 1.0f / r;
  #pragma unroll
  for (int k = 0; k < N_RBF; ++k)
    rbf[(size_t)e * N_RBF + k] = __sinf((float)(k + 1) * (PI_F / R_CUT) * r) * inv;
}

// ---------------------------------------------------------------------------
// Edge message kernel: thread j handles channel j of (d_v, d_s, d_rep).
// rbf_fcut computed on the fly (20x384 weights held in registers).
// Reads v_prev snapshot; atomically accumulates into s and v.
// ---------------------------------------------------------------------------
__global__ void __launch_bounds__(128)
edge_kernel(const int* __restrict__ edges, const float* __restrict__ rnorm,
            const float* __restrict__ rbf, const float* __restrict__ fc,
            const float* __restrict__ rbf_w,  // (20, 384) slice
            const float* __restrict__ rbf_b,  // (384) slice
            const float* __restrict__ s_pass, // (N_NODES, 384)
            const float* __restrict__ v_prev, // (N_NODES, 3, 128)
            float* __restrict__ s,            // (N_NODES, 128)
            float* __restrict__ v,            // (N_NODES, 3, 128)
            int n_edges) {
  const int j = threadIdx.x;
  float wv[N_RBF], wsc[N_RBF], wr[N_RBF];
  #pragma unroll
  for (int k = 0; k < N_RBF; ++k) {
    wv[k]  = rbf_w[k * 384 + j];
    wsc[k] = rbf_w[k * 384 + 128 + j];
    wr[k]  = rbf_w[k * 384 + 256 + j];
  }
  const float bv = rbf_b[j], bs = rbf_b[128 + j], br = rbf_b[256 + j];

  for (int e = blockIdx.x; e < n_edges; e += gridDim.x) {
    const int dst = edges[2 * e + 0];
    const int src = edges[2 * e + 1];
    const float f = fc[e];
    float av = bv, as_ = bs, ar = br;
    #pragma unroll
    for (int k = 0; k < N_RBF; ++k) {
      const float rv = rbf[(size_t)e * N_RBF + k];  // broadcast across lanes
      av += rv * wv[k]; as_ += rv * wsc[k]; ar += rv * wr[k];
    }
    av *= f; as_ *= f; ar *= f;

    const float* sp = s_pass + (size_t)src * 384;
    const float dv  = av * sp[j];
    const float dss = as_ * sp[128 + j];
    const float dr  = ar * sp[256 + j];

    atomicAdd(&s[(size_t)dst * DD + j], dss);

    const float* vs = v_prev + (size_t)src * 384;
    float* vd = v + (size_t)dst * 384;
    #pragma unroll
    for (int d = 0; d < 3; ++d) {
      const float rn = rnorm[(size_t)e * 3 + d];
      atomicAdd(&vd[d * DD + j], vs[d * DD + j] * dv + rn * dr);
    }
  }
}

// sv = [ ||V_v||_axis1 , s ]  -> (N_NODES, 256)
__global__ void vnorm_concat_kernel(const float* __restrict__ Vv,
                                    const float* __restrict__ s,
                                    float* __restrict__ sv, int n_nodes) {
  const int idx = blockIdx.x * blockDim.x + threadIdx.x;
  if (idx >= n_nodes * DD) return;
  const int node = idx >> 7, j = idx & 127;
  const float x0 = Vv[(size_t)node * 384 + j];
  const float x1 = Vv[(size_t)node * 384 + 128 + j];
  const float x2 = Vv[(size_t)node * 384 + 256 + j];
  sv[(size_t)node * 256 + j] = sqrtf(x0 * x0 + x1 * x1 + x2 * x2);
  sv[(size_t)node * 256 + 128 + j] = s[idx];
}

// s += sum_d(U_v*V_v)*asv + ass ;  v += avv * U_v
__global__ void node_update_kernel(const float* __restrict__ Uv,
                                   const float* __restrict__ Vv,
                                   const float* __restrict__ a,
                                   float* __restrict__ s, float* __restrict__ v,
                                   int n_nodes) {
  const int idx = blockIdx.x * blockDim.x + threadIdx.x;
  if (idx >= n_nodes * DD) return;
  const int node = idx >> 7, j = idx & 127;
  const float avv = a[(size_t)node * 384 + j];
  const float asv = a[(size_t)node * 384 + 128 + j];
  const float ass = a[(size_t)node * 384 + 256 + j];
  float dot = 0.0f;
  #pragma unroll
  for (int d = 0; d < 3; ++d) {
    const size_t o = (size_t)node * 384 + d * DD + j;
    const float u = Uv[o];
    dot += u * Vv[o];
    v[o] += avv * u;
  }
  s[idx] += dot * asv + ass;
}

// Final: per-node dot(h, blue_w2)+b2, atomic segment-sum over graphs.
__global__ void __launch_bounds__(128)
blue_reduce_kernel(const float* __restrict__ h, const float* __restrict__ w2,
                   const float* __restrict__ b2, const int* __restrict__ gidx,
                   float* __restrict__ out, int n_nodes) {
  const int lane = threadIdx.x & 31;
  const int wave = threadIdx.x >> 5;
  const int node = blockIdx.x * 4 + wave;
  if (node >= n_nodes) return;
  float sum = 0.0f;
  #pragma unroll
  for (int i = 0; i < 4; ++i) {
    const int j = lane + 32 * i;
    sum += h[(size_t)node * DD + j] * w2[j];
  }
  #pragma unroll
  for (int off = 16; off > 0; off >>= 1)
    sum += __shfl_down(sum, off, 32);
  if (lane == 0)
    atomicAdd(&out[gidx[node]], sum + b2[0]);
}

__global__ void zero_kernel(float* __restrict__ p, int n) {
  const int i = blockIdx.x * blockDim.x + threadIdx.x;
  if (i < n) p[i] = 0.0f;
}

// ---------------------------------------------------------------------------
extern "C" void kernel_launch(void* const* d_in, const int* in_sizes, int n_in,
                              void* d_out, int out_size, void* d_ws, size_t ws_size,
                              hipStream_t stream) {
  (void)in_sizes; (void)n_in; (void)ws_size;

  const int*   z      = (const int*)d_in[0];
  const int*   edges  = (const int*)d_in[1];
  const float* r_ij   = (const float*)d_in[2];
  const float* rnorm  = (const float*)d_in[3];
  const int*   gidx   = (const int*)d_in[4];
  const float* emb    = (const float*)d_in[5];
  const float* msg_w1 = (const float*)d_in[6];
  const float* msg_b1 = (const float*)d_in[7];
  const float* msg_w2 = (const float*)d_in[8];
  const float* msg_b2 = (const float*)d_in[9];
  const float* rbf_w  = (const float*)d_in[10];
  const float* rbf_b  = (const float*)d_in[11];
  const float* upd_U  = (const float*)d_in[12];
  const float* upd_V  = (const float*)d_in[13];
  const float* upd_w1 = (const float*)d_in[14];
  const float* upd_b1 = (const float*)d_in[15];
  const float* upd_w2 = (const float*)d_in[16];
  const float* upd_b2 = (const float*)d_in[17];
  const float* blue_w1 = (const float*)d_in[18];
  const float* blue_b1 = (const float*)d_in[19];
  const float* blue_w2 = (const float*)d_in[20];
  const float* blue_b2 = (const float*)d_in[21];
  float* out = (float*)d_out;

  // -------- workspace partition: packed f16 weights first, then f32 pool ----
  _Float16* q = (_Float16*)d_ws;
  _Float16* P_mw1 = q; q += (size_t)3 * DD * DD;
  _Float16* P_mw2 = q; q += (size_t)3 * DD * 3 * DD;
  _Float16* P_uU  = q; q += (size_t)3 * DD * DD;
  _Float16* P_uV  = q; q += (size_t)3 * DD * DD;
  _Float16* P_uw1 = q; q += (size_t)3 * 2 * DD * DD;
  _Float16* P_uw2 = q; q += (size_t)3 * DD * 3 * DD;
  _Float16* P_bw1 = q; q += (size_t)DD * DD;

  float* p     = (float*)q;
  float* s     = p; p += (size_t)N_NODES * DD;
  float* v     = p; p += (size_t)N_NODES * 3 * DD;
  float* vprev = p; p += (size_t)N_NODES * 3 * DD;
  float* h     = p; p += (size_t)N_NODES * DD;
  float* spass = p; p += (size_t)N_NODES * 3 * DD;   // later aliased as 'a'
  float* Uv    = p; p += (size_t)N_NODES * 3 * DD;
  float* Vv    = p; p += (size_t)N_NODES * 3 * DD;
  float* sv    = p; p += (size_t)N_NODES * 2 * DD;
  float* rbf   = p; p += (size_t)N_EDGES * N_RBF;
  float* fcb   = p; p += (size_t)N_EDGES;
  float* abuf  = spass;

  const int mt_s = N_NODES / 16;          // 3125 (exact)
  const int mt_v = (N_NODES * 3) / 16;    // 9375 (exact)

  auto pack = [&](const float* B, _Float16* Bp, int K, int N) {
    pack_b_kernel<<<(K * N + 255) / 256, 256, 0, stream>>>(B, Bp, K, N);
  };

  // pack all weight matrices into WMMA lane layout (f16)
  for (int i = 0; i < 3; ++i) {
    pack(msg_w1 + (size_t)i * DD * DD,         P_mw1 + (size_t)i * DD * DD,         DD, DD);
    pack(msg_w2 + (size_t)i * DD * 3 * DD,     P_mw2 + (size_t)i * DD * 3 * DD,     DD, 3 * DD);
    pack(upd_U  + (size_t)i * DD * DD,         P_uU  + (size_t)i * DD * DD,         DD, DD);
    pack(upd_V  + (size_t)i * DD * DD,         P_uV  + (size_t)i * DD * DD,         DD, DD);
    pack(upd_w1 + (size_t)i * 2 * DD * DD,     P_uw1 + (size_t)i * 2 * DD * DD,     2 * DD, DD);
    pack(upd_w2 + (size_t)i * DD * 3 * DD,     P_uw2 + (size_t)i * DD * 3 * DD,     DD, 3 * DD);
  }
  pack(blue_w1, P_bw1, DD, DD);

  zero_kernel<<<(out_size + 255) / 256, 256, 0, stream>>>(out, out_size);
  init_nodes_kernel<<<(N_NODES * 3 * DD + 255) / 256, 256, 0, stream>>>(z, emb, s, v, N_NODES);
  init_edges_kernel<<<(N_EDGES + 127) / 128, 128, 0, stream>>>(r_ij, rbf, fcb, N_EDGES);

  for (int i = 0; i < 3; ++i) {
    const float* mb1 = msg_b1 + (size_t)i * DD;
    const float* mb2 = msg_b2 + (size_t)i * 3 * DD;
    const float* rw  = rbf_w  + (size_t)i * N_RBF * 3 * DD;
    const float* rb  = rbf_b  + (size_t)i * 3 * DD;
    const float* ub1 = upd_b1 + (size_t)i * DD;
    const float* ub2 = upd_b2 + (size_t)i * 3 * DD;

    // message MLP on nodes
    wmma_gemm_kernel<1, 128><<<dim3(mt_s, 1), 256, 0, stream>>>(
        s, P_mw1 + (size_t)i * DD * DD, mb1, h, N_NODES, DD);
    wmma_gemm_kernel<0, 128><<<dim3(mt_s, 3), 256, 0, stream>>>(
        h, P_mw2 + (size_t)i * DD * 3 * DD, mb2, spass, N_NODES, 3 * DD);

    // snapshot v (edge phase reads old v, scatters into new v)
    hipMemcpyAsync(vprev, v, (size_t)N_NODES * 3 * DD * sizeof(float),
                   hipMemcpyDeviceToDevice, stream);
    edge_kernel<<<4096, 128, 0, stream>>>(edges, rnorm, rbf, fcb, rw, rb,
                                          spass, vprev, s, v, N_EDGES);

    // equivariant update
    wmma_gemm_kernel<0, 128><<<dim3(mt_v, 1), 256, 0, stream>>>(
        v, P_uU + (size_t)i * DD * DD, nullptr, Uv, N_NODES * 3, DD);
    wmma_gemm_kernel<0, 128><<<dim3(mt_v, 1), 256, 0, stream>>>(
        v, P_uV + (size_t)i * DD * DD, nullptr, Vv, N_NODES * 3, DD);
    vnorm_concat_kernel<<<(N_NODES * DD + 255) / 256, 256, 0, stream>>>(Vv, s, sv, N_NODES);
    wmma_gemm_kernel<1, 256><<<dim3(mt_s, 1), 256, 0, stream>>>(
        sv, P_uw1 + (size_t)i * 2 * DD * DD, ub1, h, N_NODES, DD);
    wmma_gemm_kernel<0, 128><<<dim3(mt_s, 3), 256, 0, stream>>>(
        h, P_uw2 + (size_t)i * DD * 3 * DD, ub2, abuf, N_NODES, 3 * DD);
    node_update_kernel<<<(N_NODES * DD + 255) / 256, 256, 0, stream>>>(Uv, Vv, abuf, s, v, N_NODES);
  }

  // readout
  wmma_gemm_kernel<1, 128><<<dim3(mt_s, 1), 256, 0, stream>>>(
      s, P_bw1, blue_b1, h, N_NODES, DD);
  blue_reduce_kernel<<<(N_NODES + 3) / 4, 128, 0, stream>>>(h, blue_w2, blue_b2, gidx, out, N_NODES);
}
